// CausalSelfAttention_4440996184360
// MI455X (gfx1250) — matrix operand order repaired and checked by reading, not compiled, not run on hardware
//
#include <hip/hip_runtime.h>

typedef __attribute__((ext_vector_type(16))) _Float16 v16h;
typedef __attribute__((ext_vector_type(8)))  _Float16 v8h;
typedef __attribute__((ext_vector_type(8)))  float    v8f;

// Problem dims (match reference setup_inputs)
constexpr int Bsz  = 4;
constexpr int Tseq = 2048;
constexpr int C    = 1024;
constexpr int H    = 16;
constexpr int Dh   = 64;          // C / H
constexpr int Mrows = Bsz * Tseq; // 8192

// ---------------------------------------------------------------------------
// WMMA helpers (CDNA5 wave32, 16x16x32 f16 -> f32)
// ---------------------------------------------------------------------------
__device__ __forceinline__ v8f wmma16(v16h a, v16h b, v8f c) {
  // (neg_a, A, neg_b, B, c_mod, C, reuse_a, reuse_b)
  return __builtin_amdgcn_wmma_f32_16x16x32_f16(false, a, false, b, (short)0, c,
                                                false, false);
}

// A-matrix fragment (16 M x 32 K), row-major source, K contiguous.
// lane L: row = L&15 ; halves 0..7 = K = sel*8+{0..7}; halves 8..15 = K = 16+sel*8+{0..7}
__device__ __forceinline__ v16h load_a_frag(const _Float16* A, int ld, int row0,
                                            int k0, int lane) {
  const int m    = lane & 15;
  const int sel8 = (lane >> 4) * 8;
  const _Float16* p = A + (size_t)(row0 + m) * ld + k0 + sel8;
  v8h lo = *(const v8h*)(p);
  v8h hi = *(const v8h*)(p + 16);
  v16h r;
#pragma unroll
  for (int i = 0; i < 8; ++i) { r[i] = lo[i]; r[i + 8] = hi[i]; }
  return r;
}

// B-matrix fragment (32 K x 16 N) where GEMM-B column n is a K-contiguous row
// of the source matrix Bm[N][K] (NT layout: y = A * Bm^T).
// lane L: col = L&15 ; halves 0..7 = K = k0 + (L>>4)*8 + {0..7}; halves 8..15 = K = k0 + 16 + (L>>4)*8 + {0..7}
__device__ __forceinline__ v16h load_b_frag(const _Float16* Bm, int ld, int col0,
                                            int k0, int lane) {
  const int n     = lane & 15;
  const int sel8  = (lane >> 4) * 8;
  const _Float16* p = Bm + (size_t)(col0 + n) * ld + k0 + sel8;
  v8h lo = *(const v8h*)(p);
  v8h hi = *(const v8h*)(p + 16);
  v16h r;
#pragma unroll
  for (int i = 0; i < 8; ++i) { r[i] = lo[i]; r[i + 8] = hi[i]; }
  return r;
}

__device__ __forceinline__ float rowmax16(float v) {
  v = fmaxf(v, __shfl_xor(v, 1, 32));
  v = fmaxf(v, __shfl_xor(v, 2, 32));
  v = fmaxf(v, __shfl_xor(v, 4, 32));
  v = fmaxf(v, __shfl_xor(v, 8, 32));
  return v;
}
__device__ __forceinline__ float rowsum16(float v) {
  v += __shfl_xor(v, 1, 32);
  v += __shfl_xor(v, 2, 32);
  v += __shfl_xor(v, 4, 32);
  v += __shfl_xor(v, 8, 32);
  return v;
}

// ---------------------------------------------------------------------------
// fp32 -> f16 conversion (grid-stride)
// ---------------------------------------------------------------------------
__global__ void cvt_f16(const float* __restrict__ in, _Float16* __restrict__ out,
                        int n) {
  int i = blockIdx.x * blockDim.x + threadIdx.x;
  int stride = gridDim.x * blockDim.x;
  for (; i < n; i += stride) out[i] = (_Float16)in[i];
}

// ---------------------------------------------------------------------------
// Fused QKV projection: y = x @ W^T + b  (NT GEMM, M=8192, N=1024, K=1024)
// blockIdx.z selects Q/K/V. Q,K stored [B,H,T,D]; V stored transposed [B,H,D,T].
// Block = 4 waves; wave tile 32x64; block tile 64x128.
// ---------------------------------------------------------------------------
__global__ __launch_bounds__(128) void qkv_gemm(
    const _Float16* __restrict__ X,
    const _Float16* __restrict__ Wq16, const _Float16* __restrict__ Wk16,
    const _Float16* __restrict__ Wv16,
    const float* __restrict__ bq, const float* __restrict__ bk,
    const float* __restrict__ bv,
    _Float16* __restrict__ Qo, _Float16* __restrict__ Ko,
    _Float16* __restrict__ Vt) {
  const int lane = threadIdx.x & 31;
  const int wv   = threadIdx.x >> 5;
  const int m0   = blockIdx.x * 64 + (wv & 1) * 32;
  const int n0   = blockIdx.y * 128 + (wv >> 1) * 64;
  const int z    = blockIdx.z;

  const _Float16* W = (z == 0) ? Wq16 : ((z == 1) ? Wk16 : Wv16);
  const float* bias = (z == 0) ? bq : ((z == 1) ? bk : bv);

  v8f acc[2][4] = {};
  for (int k0 = 0; k0 < C; k0 += 32) {
    v16h a0 = load_a_frag(X, C, m0, k0, lane);
    v16h a1 = load_a_frag(X, C, m0 + 16, k0, lane);
#pragma unroll
    for (int jn = 0; jn < 4; ++jn) {
      v16h bf = load_b_frag(W, C, n0 + jn * 16, k0, lane);
      acc[0][jn] = wmma16(a0, bf, acc[0][jn]);
      acc[1][jn] = wmma16(a1, bf, acc[1][jn]);
    }
  }

  const int nlane = lane & 15;
  const int sel8  = (lane >> 4) * 8;
#pragma unroll
  for (int im = 0; im < 2; ++im)
#pragma unroll
    for (int jn = 0; jn < 4; ++jn)
#pragma unroll
      for (int i = 0; i < 8; ++i) {
        const int m = m0 + im * 16 + sel8 + i;
        const int n = n0 + jn * 16 + nlane;
        const float val = acc[im][jn][i] + bias[n];
        const int bb = m >> 11, t = m & (Tseq - 1);
        const int hh = n >> 6, d = n & (Dh - 1);
        if (z == 0)
          Qo[(((size_t)bb * H + hh) * Tseq + t) * Dh + d] = (_Float16)val;
        else if (z == 1)
          Ko[(((size_t)bb * H + hh) * Tseq + t) * Dh + d] = (_Float16)val;
        else
          Vt[(((size_t)bb * H + hh) * Dh + d) * Tseq + t] = (_Float16)val;
      }
}

// ---------------------------------------------------------------------------
// Flash attention (causal, online softmax). One wave owns 16 q-rows; block of
// 4 waves covers 64 q-rows. Key blocks of 32. Q,K: [B,H,T,D]; Vt: [B,H,D,T].
// Output att: [B,T,C] f16 (heads merged), ready for the NT output projection.
// ---------------------------------------------------------------------------
__global__ __launch_bounds__(128) void attn_kernel(
    const _Float16* __restrict__ Q, const _Float16* __restrict__ K,
    const _Float16* __restrict__ Vt, _Float16* __restrict__ att) {
  __shared__ _Float16 Pl[4][16][32];  // per-wave P transpose bounce (1KB/wave)

  const int lane  = threadIdx.x & 31;
  const int wv    = threadIdx.x >> 5;
  const int nlane = lane & 15;
  const int sel8  = (lane >> 4) * 8;
  const int q0    = blockIdx.x * 64 + wv * 16;
  const int h     = blockIdx.y;
  const int b     = blockIdx.z;

  const size_t headoff = ((size_t)(b * H + h)) * Tseq * Dh;
  const _Float16* Qh = Q + headoff;
  const _Float16* Kh = K + headoff;
  const _Float16* Vh = Vt + headoff;  // [Dh][Tseq]

  const float scale = 0.125f;  // 1/sqrt(64)
  const float NEG   = -__builtin_huge_valf();

  v16h a_q[2];
  a_q[0] = load_a_frag(Qh, Dh, q0, 0, lane);
  a_q[1] = load_a_frag(Qh, Dh, q0, 32, lane);

  v8f acc_o[4] = {};
  float m_run[8], l_run[8];
#pragma unroll
  for (int i = 0; i < 8; ++i) { m_run[i] = NEG; l_run[i] = 0.0f; }

  for (int kb = 0; kb < q0 + 16; kb += 32) {
    // S = Q * K^T for 16 rows x 32 keys (4 WMMAs: 2 K-steps x 2 key tiles)
    v8f sa[2] = {};
#pragma unroll
    for (int kk = 0; kk < 2; ++kk)
#pragma unroll
      for (int j = 0; j < 2; ++j) {
        v16h bk = load_b_frag(Kh, Dh, kb + j * 16, kk * 32, lane);
        sa[j] = wmma16(a_q[kk], bk, sa[j]);
      }

    // scale + causal mask + online softmax (row stats via 16-lane shuffles)
#pragma unroll
    for (int i = 0; i < 8; ++i) {
      const int row = q0 + sel8 + i;
      float s0 = sa[0][i] * scale;
      float s1 = sa[1][i] * scale;
      if (kb + nlane > row) s0 = NEG;
      if (kb + 16 + nlane > row) s1 = NEG;
      const float mx    = rowmax16(fmaxf(s0, s1));
      const float m_new = fmaxf(m_run[i], mx);
      const float p0 = __expf(s0 - m_new);
      const float p1 = __expf(s1 - m_new);
      const float rs = rowsum16(p0 + p1);
      const float alpha = __expf(m_run[i] - m_new);
      m_run[i] = m_new;
      l_run[i] = l_run[i] * alpha + rs;
#pragma unroll
      for (int jd = 0; jd < 4; ++jd) acc_o[jd][i] *= alpha;
      // stash probabilities (C-layout -> LDS) for A-fragment reload
      Pl[wv][sel8 + i][nlane]      = (_Float16)p0;
      Pl[wv][sel8 + i][16 + nlane] = (_Float16)p1;
    }

    // wave-private LDS RAW: wait for all DS ops before cross-lane reload
    asm volatile("s_wait_dscnt 0x0" ::: "memory");

    // reload P as a 16x32 A-fragment
    v16h a_p;
    {
      const _Float16* pr = &Pl[wv][nlane][0];
#pragma unroll
      for (int i = 0; i < 8; ++i) {
        a_p[i]     = pr[sel8 + i];
        a_p[8 + i] = pr[16 + sel8 + i];
      }
    }

    // O += P * V  (4 WMMAs over d-tiles; Vt rows are K-contiguous)
#pragma unroll
    for (int jd = 0; jd < 4; ++jd) {
      v16h bvf = load_b_frag(Vh, Tseq, jd * 16, kb, lane);
      acc_o[jd] = wmma16(a_p, bvf, acc_o[jd]);
    }
  }

  // epilogue: normalize and store merged-head layout [B,T,C]
#pragma unroll
  for (int jd = 0; jd < 4; ++jd)
#pragma unroll
    for (int i = 0; i < 8; ++i) {
      const int row = q0 + sel8 + i;
      const float o = acc_o[jd][i] / l_run[i];
      att[((size_t)(b * Tseq + row)) * C + h * Dh + jd * 16 + nlane] =
          (_Float16)o;
    }
}

// ---------------------------------------------------------------------------
// Output projection: out = att @ Wp^T + bp (fp32 output)
// ---------------------------------------------------------------------------
__global__ __launch_bounds__(128) void proj_gemm(
    const _Float16* __restrict__ A, const _Float16* __restrict__ W,
    const float* __restrict__ bias, float* __restrict__ Y) {
  const int lane = threadIdx.x & 31;
  const int wv   = threadIdx.x >> 5;
  const int m0   = blockIdx.x * 64 + (wv & 1) * 32;
  const int n0   = blockIdx.y * 128 + (wv >> 1) * 64;

  v8f acc[2][4] = {};
  for (int k0 = 0; k0 < C; k0 += 32) {
    v16h a0 = load_a_frag(A, C, m0, k0, lane);
    v16h a1 = load_a_frag(A, C, m0 + 16, k0, lane);
#pragma unroll
    for (int jn = 0; jn < 4; ++jn) {
      v16h bf = load_b_frag(W, C, n0 + jn * 16, k0, lane);
      acc[0][jn] = wmma16(a0, bf, acc[0][jn]);
      acc[1][jn] = wmma16(a1, bf, acc[1][jn]);
    }
  }

  const int nlane = lane & 15;
  const int sel8  = (lane >> 4) * 8;
#pragma unroll
  for (int im = 0; im < 2; ++im)
#pragma unroll
    for (int jn = 0; jn < 4; ++jn)
#pragma unroll
      for (int i = 0; i < 8; ++i) {
        const int m = m0 + im * 16 + sel8 + i;
        const int n = n0 + jn * 16 + nlane;
        Y[(size_t)m * C + n] = acc[im][jn][i] + bias[n];
      }
}

// ---------------------------------------------------------------------------
// Host launcher
// ---------------------------------------------------------------------------
extern "C" void kernel_launch(void* const* d_in, const int* in_sizes, int n_in,
                              void* d_out, int out_size, void* d_ws,
                              size_t ws_size, hipStream_t stream) {
  const float* x  = (const float*)d_in[0];
  const float* Wq = (const float*)d_in[1];
  const float* bq = (const float*)d_in[2];
  const float* Wk = (const float*)d_in[3];
  const float* bk = (const float*)d_in[4];
  const float* Wv = (const float*)d_in[5];
  const float* bv = (const float*)d_in[6];
  const float* Wp = (const float*)d_in[7];
  const float* bp = (const float*)d_in[8];
  float* out = (float*)d_out;

  // Workspace layout (bytes). Total = 88 MB.
  char* ws = (char*)d_ws;
  const size_t XC  = (size_t)Mrows * C;  // 8,388,608 elems
  const size_t WC  = (size_t)C * C;      // 1,048,576 elems
  _Float16* x16   = (_Float16*)(ws);
  _Float16* Wq16  = (_Float16*)(ws + 2 * XC);
  _Float16* Wk16  = (_Float16*)(ws + 2 * (XC + 1 * WC));
  _Float16* Wv16  = (_Float16*)(ws + 2 * (XC + 2 * WC));
  _Float16* Wp16  = (_Float16*)(ws + 2 * (XC + 3 * WC));
  _Float16* Q16   = (_Float16*)(ws + 2 * (XC + 4 * WC));
  _Float16* K16   = (_Float16*)(ws + 2 * (2 * XC + 4 * WC));
  _Float16* Vt16  = (_Float16*)(ws + 2 * (3 * XC + 4 * WC));
  _Float16* att16 = (_Float16*)(ws + 2 * (4 * XC + 4 * WC));

  // 1) fp32 -> f16 conversions
  cvt_f16<<<2048, 256, 0, stream>>>(x, x16, (int)XC);
  cvt_f16<<<512, 256, 0, stream>>>(Wq, Wq16, (int)WC);
  cvt_f16<<<512, 256, 0, stream>>>(Wk, Wk16, (int)WC);
  cvt_f16<<<512, 256, 0, stream>>>(Wv, Wv16, (int)WC);
  cvt_f16<<<512, 256, 0, stream>>>(Wp, Wp16, (int)WC);

  // 2) fused QKV projection (z: 0=Q, 1=K, 2=V-transposed)
  qkv_gemm<<<dim3(Mrows / 64, C / 128, 3), 128, 0, stream>>>(
      x16, Wq16, Wk16, Wv16, bq, bk, bv, Q16, K16, Vt16);

  // 3) causal flash attention
  attn_kernel<<<dim3(Tseq / 64, H, Bsz), 128, 0, stream>>>(Q16, K16, Vt16,
                                                           att16);

  // 4) output projection -> fp32 d_out
  proj_gemm<<<dim3(Mrows / 64, C / 128), 128, 0, stream>>>(att16, Wp16, bp,
                                                           out);
}
